// VectorQuantizer_53266184405827
// MI455X (gfx1250) — matrix operand order repaired and checked by reading, not compiled, not run on hardware
//
#include <hip/hip_runtime.h>

// VQ-VAE vector quantizer for MI455X (gfx1250, wave32, WMMA).
// Needs ws_size >= ~9 MB:
//   [0,4MB)            z as f16, (BT=8192, D=256) row-major
//   [4MB,8MB)          codebook as f16, (K=8192, D=256) row-major
//   [8MB,+32KB)        ||c_k||^2 fp32
//   [+32KB,+64KB)      argmin indices (int32)
//   [+64KB,+72KB)      per-block loss partials (2048 floats)
//   [+72KB,+328KB)     partial argmin scores (8192 x 8 floats)
//   [+328KB,+584KB)    partial argmin indices (8192 x 8 int32)

typedef __attribute__((ext_vector_type(16))) _Float16 v16h;
typedef __attribute__((ext_vector_type(8)))  _Float16 v8h;
typedef __attribute__((ext_vector_type(8)))  float    v8f;

#define B_  8
#define D_  256
#define T_  1024
#define K_  8192
#define BT_ (B_ * T_)
#define NCHUNK 8                          // N-dim split for occupancy
#define NT_PER_CHUNK (K_ / 16 / NCHUNK)   // 64 N-tiles per chunk

// ---------------------------------------------------------------- kernel 1
// Convert codebook to f16 and compute ||c_k||^2 in fp32.
__global__ void vq_prep_codebook(const float* __restrict__ cb,
                                 _Float16* __restrict__ ch,
                                 float* __restrict__ c2) {
    const int k = blockIdx.x;
    const int d = threadIdx.x;
    const float v = cb[k * D_ + d];
    ch[k * D_ + d] = (_Float16)v;
    __shared__ float red[256];
    red[d] = v * v;
    __syncthreads();
    for (int off = 128; off > 0; off >>= 1) {
        if (d < off) red[d] += red[d + off];
        __syncthreads();
    }
    if (d == 0) c2[k] = red[0];
}

// ---------------------------------------------------------------- kernel 2
// Transpose-convert z: (B, D, T) fp32 -> (B*T, D) f16, 32x32 LDS tiles so
// both the global read (along t) and write (along d) are coalesced.
__global__ void vq_prep_z(const float* __restrict__ z,
                          _Float16* __restrict__ zh) {
    const int blk = blockIdx.x;          // 8 * 8 * 32 = 2048 blocks
    const int b   = blk >> 8;
    const int rem = blk & 255;
    const int dt  = rem >> 5;            // 8 d-tiles
    const int tt  = rem & 31;            // 32 t-tiles
    const int d0 = dt * 32, t0 = tt * 32;
    __shared__ float tile[32][33];
    const int tid = threadIdx.x;
#pragma unroll
    for (int i = 0; i < 4; ++i) {
        const int e = tid + i * 256;
        const int dl = e >> 5, tl = e & 31;     // coalesced along t
        tile[dl][tl] = z[b * (D_ * T_) + (d0 + dl) * T_ + (t0 + tl)];
    }
    __syncthreads();
#pragma unroll
    for (int i = 0; i < 4; ++i) {
        const int e = tid + i * 256;
        const int tl = e >> 5, dl = e & 31;     // coalesced along d
        zh[(b * T_ + t0 + tl) * D_ + d0 + dl] = (_Float16)tile[dl][tl];
    }
}

// ---------------------------------------------------------------- kernel 3
// WMMA partial argmin. score_k = ||c_k||^2 - 2 z.c_k  (||z||^2 row-constant).
// Grid: 512 blocks = 64 M-groups x 8 N-chunks; 8 waves/block, one wave per
// 16-row M-tile. A slice register-resident (8 x v16h). B fragments are
// double-buffered across N-tiles: the next tile's 16 global_load_b128 issue
// before the current tile's 8 WMMAs, so a full tile of matrix work covers the
// load latency. Tail overrun past the last tile stays inside the workspace.
__global__ void __launch_bounds__(256)
vq_argmin_part(const _Float16* __restrict__ zh,
               const _Float16* __restrict__ ch,
               const float* __restrict__ c2,
               float* __restrict__ pscore,
               int* __restrict__ pidx) {
    const int lane  = threadIdx.x & 31;
    const int wave  = threadIdx.x >> 5;
    const int chunk = blockIdx.x & (NCHUNK - 1);
    const int mt    = (blockIdx.x >> 3) * 8 + wave;   // 0..511 M-tiles
    const int khalf = lane >> 4;                 // A: K 0-15 vs 16-31
    const int arow  = (mt << 4) + (lane & 15);   // A-layout row for this lane

    // Preload A fragments: per lane one contiguous 32B chunk per k-step of 32.
    v16h a[8];
    const _Float16* zrow = zh + arow * D_;
#pragma unroll
    for (int s = 0; s < 8; ++s) {
        a[s] = *(const v16h*)(zrow + s * 32 + khalf * 16);
    }

    float best[8];
    int   bidx[8];
#pragma unroll
    for (int j = 0; j < 8; ++j) { best[j] = 3.4e38f; bidx[j] = 0; }

    const int ncol = lane & 15;                  // B/C column for this lane
    const int nt0  = chunk * NT_PER_CHUNK;

    // B 32x16, k order matching A: lane half selects K 0-15 vs 16-31 -> one contiguous
    // 32B chunk of codebook row `code` per k-step.
    const _Float16* crow = ch + ((nt0 << 4) + ncol) * D_ + khalf * 16;

    // Prologue: load tile nt0's fragments.
    v16h bf[8];
#pragma unroll
    for (int s = 0; s < 8; ++s)
        bf[s] = *(const v16h*)(crow + s * 32);

    for (int nt = nt0; nt < nt0 + NT_PER_CHUNK; ++nt) {
        const int code = (nt << 4) + ncol;
        const _Float16* cnext = crow + (D_ << 4);       // next tile's row
        __builtin_prefetch(cnext + (D_ << 4), 0, 3);    // 2 tiles ahead, WGP scope
        const float c2v = c2[code];

        // Issue next tile's loads before consuming this tile's fragments.
        v16h bn[8];
#pragma unroll
        for (int s = 0; s < 8; ++s)
            bn[s] = *(const v16h*)(cnext + s * 32);

        v8f acc = {};
#pragma unroll
        for (int s = 0; s < 8; ++s)
            acc = __builtin_amdgcn_wmma_f32_16x16x32_f16(
                false, a[s], false, bf[s], (short)0, acc, false, false);

#pragma unroll
        for (int j = 0; j < 8; ++j) {
            const float sc = __builtin_fmaf(-2.0f, acc[j], c2v);
            if (sc < best[j]) { best[j] = sc; bidx[j] = code; }  // first idx wins
        }

#pragma unroll
        for (int s = 0; s < 8; ++s)
            bf[s] = bn[s];
        crow = cnext;
    }

    // Reduce across the 16 column-lanes of each half-wave (tie -> lower idx).
#pragma unroll
    for (int j = 0; j < 8; ++j) {
        float s = best[j];
        int   bi = bidx[j];
#pragma unroll
        for (int off = 1; off < 16; off <<= 1) {
            const float s2 = __shfl_xor(s, off, 32);
            const int   i2 = __shfl_xor(bi, off, 32);
            if (s2 < s || (s2 == s && i2 < bi)) { s = s2; bi = i2; }
        }
        if (ncol == 0) {
            const int row = (mt << 4) + j + (khalf << 3);  // C-layout row
            pscore[row * NCHUNK + chunk] = s;
            pidx[row * NCHUNK + chunk]   = bi;
        }
    }
}

// ---------------------------------------------------------------- kernel 4
// Merge the 8 per-chunk partials per row; fixed ascending chunk order +
// strict < keeps the lowest winning index (argmin semantics). Deterministic.
__global__ void vq_merge(const float* __restrict__ pscore,
                         const int* __restrict__ pidx,
                         int* __restrict__ idx_ws,
                         float* __restrict__ out_idx) {
    const int row = blockIdx.x * 256 + threadIdx.x;
    float s  = pscore[row * NCHUNK];
    int   bi = pidx[row * NCHUNK];
#pragma unroll
    for (int c = 1; c < NCHUNK; ++c) {
        const float s2 = pscore[row * NCHUNK + c];
        const int   i2 = pidx[row * NCHUNK + c];
        if (s2 < s || (s2 == s && i2 < bi)) { s = s2; bi = i2; }
    }
    idx_ws[row]  = bi;
    out_idx[row] = (float)bi;
}

// ---------------------------------------------------------------- kernel 5
// Gather fp32 codebook rows by index, write z_q (== straight-through forward
// value) in (B, D, T) layout via LDS transpose; per-block sum of (z_q - z)^2.
__global__ void vq_gather(const float* __restrict__ z,
                          const float* __restrict__ cb,
                          const int* __restrict__ idx_ws,
                          float* __restrict__ out_zq,
                          float* __restrict__ blocksum) {
    const int blk = blockIdx.x;          // 8 * 32 * 8 = 2048 blocks
    const int b   = blk >> 8;
    const int rem = blk & 255;
    const int tt  = rem >> 3;            // 32 t-tiles
    const int dt  = rem & 7;             // 8 d-tiles
    const int t0 = tt * 32, d0 = dt * 32;
    __shared__ float tile[32][33];
    __shared__ int   sidx[32];
    const int tid = threadIdx.x;
    if (tid < 32) sidx[tid] = idx_ws[b * T_ + t0 + tid];
    __syncthreads();
#pragma unroll
    for (int i = 0; i < 4; ++i) {
        const int e = tid + i * 256;
        const int tl = e >> 5, dl = e & 31;           // coalesced along d
        tile[tl][dl] = cb[sidx[tl] * D_ + d0 + dl];
    }
    __syncthreads();
    float part = 0.0f;
#pragma unroll
    for (int i = 0; i < 4; ++i) {
        const int e = tid + i * 256;
        const int dl = e >> 5, tl = e & 31;           // coalesced along t
        const float c = tile[tl][dl];
        const int o = b * (D_ * T_) + (d0 + dl) * T_ + t0 + tl;
        const float zv = z[o];
        out_zq[o] = c;
        const float dif = c - zv;
        part += dif * dif;
    }
    __shared__ float red[256];
    red[tid] = part;
    __syncthreads();
    for (int off = 128; off > 0; off >>= 1) {
        if (tid < off) red[tid] += red[tid + off];
        __syncthreads();
    }
    if (tid == 0) blocksum[blk] = red[0];
}

// ---------------------------------------------------------------- kernel 6
// Deterministic fixed-order reduction of 2048 partials.
// loss = (1 + 0.25) * mean((z_q - z)^2)
__global__ void vq_finalize(const float* __restrict__ blocksum,
                            float* __restrict__ out_loss) {
    __shared__ float red[256];
    const int tid = threadIdx.x;
    float s = 0.0f;
    for (int i = tid; i < 2048; i += 256) s += blocksum[i];
    red[tid] = s;
    __syncthreads();
    for (int off = 128; off > 0; off >>= 1) {
        if (tid < off) red[tid] += red[tid + off];
        __syncthreads();
    }
    if (tid == 0)
        out_loss[0] = red[0] * (1.25f / (float)(B_ * D_ * T_));
}

// ----------------------------------------------------------------
extern "C" void kernel_launch(void* const* d_in, const int* in_sizes, int n_in,
                              void* d_out, int out_size, void* d_ws, size_t ws_size,
                              hipStream_t stream) {
    (void)in_sizes; (void)n_in; (void)out_size; (void)ws_size;
    const float* z  = (const float*)d_in[0];   // (8, 256, 1024) fp32
    const float* cb = (const float*)d_in[1];   // (8192, 256) fp32

    float* out      = (float*)d_out;
    float* out_zq   = out;                       // 2,097,152 floats (B,D,T)
    float* out_idx  = out + 2097152;             // 8,192 floats (B,T)
    float* out_loss = out + 2097152 + 8192;      // 1 float

    char* ws = (char*)d_ws;
    _Float16* zh    = (_Float16*)(ws);
    _Float16* ch    = (_Float16*)(ws + (4u << 20));
    float*    c2    = (float*)(ws + (8u << 20));
    int*      idxw  = (int*)(ws + (8u << 20) + (32u << 10));
    float*    bsum  = (float*)(ws + (8u << 20) + (64u << 10));
    float*    pscr  = (float*)(ws + (8u << 20) + (72u << 10));
    int*      pidx  = (int*)(ws + (8u << 20) + (328u << 10));

    vq_prep_codebook<<<K_, 256, 0, stream>>>(cb, ch, c2);
    vq_prep_z<<<2048, 256, 0, stream>>>(z, zh);
    vq_argmin_part<<<(BT_ / 16 / 8) * NCHUNK, 256, 0, stream>>>(zh, ch, c2, pscr, pidx);
    vq_merge<<<BT_ / 256, 256, 0, stream>>>(pscr, pidx, idxw, out_idx);
    vq_gather<<<2048, 256, 0, stream>>>(z, cb, idxw, out_zq, bsum);
    vq_finalize<<<1, 256, 0, stream>>>(bsum, out_loss);
}
